// GroupedAttentionCore_40175124087336
// MI455X (gfx1250) — matrix operand order repaired and checked by reading, not compiled, not run on hardware
//
#include <hip/hip_runtime.h>
#include <hip/hip_bf16.h>
#include <math.h>

// ---- problem constants (from reference) ----
#define HQ_   32
#define HKV_  8
#define DQK_  64
#define DV_   64
#define HID_  2048
#define GRP_  4
#define BATCH_ 2
#define SEQ_  2048
#define QSTR_ (HQ_ * DQK_)   // 2048 floats per Q row
#define KSTR_ (HKV_ * DQK_)  // 512
#define VSTR_ (HKV_ * DV_)   // 512
#define OSTR_ (HQ_ * DV_)    // 2048 (attention output feature dim)
#define SCALE_ 0.125f        // 1/sqrt(64)

typedef __attribute__((ext_vector_type(16))) __bf16 v16bf;
typedef __attribute__((ext_vector_type(2)))  __bf16 v2bf;
typedef __attribute__((ext_vector_type(8)))  float  v8f;

union BFrag {
  v16bf v;
  unsigned u[8];
  uint4 q[2];
};

#if defined(__has_builtin)
#  if __has_builtin(__builtin_amdgcn_cvt_pk_bf16_f32)
#    define PK_BF16_HW 1
#  endif
#endif

// pack two f32 -> two bf16 in one dword (HW packed convert when available)
__device__ __forceinline__ unsigned pk2(float lo, float hi) {
#ifdef PK_BF16_HW
  v2bf t = __builtin_amdgcn_cvt_pk_bf16_f32(lo, hi);
  return __builtin_bit_cast(unsigned, t);
#else
  unsigned a = __float_as_uint(lo), b = __float_as_uint(hi);
  a = (a + 0x7FFFu + ((a >> 16) & 1u)) >> 16;  // RNE
  b = (b + 0x7FFFu + ((b >> 16) & 1u)) >> 16;
  return (a & 0xFFFFu) | (b << 16);
#endif
}
__device__ __forceinline__ unsigned short f2bf(float f) {
  return (unsigned short)(pk2(f, 0.f) & 0xFFFFu);
}

// =====================================================================
// Kernel 1: GQA flash attention.  grid = (SEQ/64, B*HQ), block = 128 (4 waves)
// Each wave owns a 16-row Q tile.  32-key K/V tiles are staged through a
// double-buffered LDS pipeline in exact WMMA B-fragment layout, so the
// global loads of tile i+1 overlap the WMMA stream of tile i and fragment
// reads are aligned ds_load_b128 pairs.  Output: bf16 [B*S, HQ*DV] in ws.
// =====================================================================
__global__ __launch_bounds__(128)
void gqa_attn_kernel(const float* __restrict__ Q, const float* __restrict__ K,
                     const float* __restrict__ V, unsigned short* __restrict__ Obf)
{
  // [buf][frag][lane][16 bf16] stored as 8 dwords per lane
  __shared__ __align__(16) unsigned kldsu[2][4 * 32 * 8];   // 2 x 4 KB
  __shared__ __align__(16) unsigned vldsu[2][4 * 32 * 8];   // 2 x 4 KB
  __shared__ __align__(16) unsigned short plds[4][16][32];  // 4 KB per-wave P relayout

  const int tid  = threadIdx.x;
  const int wave = tid >> 5;
  const int lane = tid & 31;
  const int lh   = lane >> 4;   // lane half selects K-subrange per ISA layout
  const int ll   = lane & 15;

  const int bh  = blockIdx.y;
  const int b   = bh / HQ_;
  const int h   = bh % HQ_;
  const int kvh = h / GRP_;
  const int q0  = blockIdx.x * 64 + wave * 16;

  // ---- Q A-fragments (16x32 bf16, two k-chunks), held in registers ----
  // A layout: lane<16 -> row=lane, K {0..7,16..23}; lane>=16 -> row=lane-16, K {8..15,24..31}
  BFrag qa[2];
  {
    const float* qrow = Q + (size_t)(b * SEQ_ + q0 + ll) * QSTR_ + h * DQK_;
#pragma unroll
    for (int c = 0; c < 2; ++c) {
      const float* qp = qrow + c * 32 + lh * 8;
      float4 f0 = *(const float4*)(qp + 0);
      float4 f1 = *(const float4*)(qp + 4);
      float4 f2 = *(const float4*)(qp + 16);
      float4 f3 = *(const float4*)(qp + 20);
      qa[c].u[0] = pk2(f0.x, f0.y); qa[c].u[1] = pk2(f0.z, f0.w);
      qa[c].u[2] = pk2(f1.x, f1.y); qa[c].u[3] = pk2(f1.z, f1.w);
      qa[c].u[4] = pk2(f2.x, f2.y); qa[c].u[5] = pk2(f2.z, f2.w);
      qa[c].u[6] = pk2(f3.x, f3.y); qa[c].u[7] = pk2(f3.z, f3.w);
    }
  }

  v8f acc0 = {}, acc1 = {}, acc2 = {}, acc3 = {};  // O accum: 16x64 f32
  float m[8], l[8];
#pragma unroll
  for (int r = 0; r < 8; ++r) { m[r] = -INFINITY; l[r] = 0.f; }

  // cooperative-loader thread mapping (fragment slot + lane within slot)
  const int lf  = tid >> 5;
  const int lln = tid & 31;
  const int llh = lln >> 4;
  const int lll = lln & 15;

  float kf[16], vf[16];  // staging registers for the in-flight tile

  // issue global loads for key tile at sb
  auto load_tile = [&](int sb) {
    const int c = lf >> 1, t = lf & 1;
    const int key = t * 16 + lll;
    const float* kp = K + (size_t)(b * SEQ_ + sb + key) * KSTR_ + kvh * DQK_ + c * 32 + llh * 8;
    float4 f0 = *(const float4*)(kp + 0);
    float4 f1 = *(const float4*)(kp + 4);
    float4 f2 = *(const float4*)(kp + 16);
    float4 f3 = *(const float4*)(kp + 20);
    kf[0]=f0.x; kf[1]=f0.y; kf[2]=f0.z; kf[3]=f0.w;
    kf[4]=f1.x; kf[5]=f1.y; kf[6]=f1.z; kf[7]=f1.w;
    kf[8]=f2.x; kf[9]=f2.y; kf[10]=f2.z; kf[11]=f2.w;
    kf[12]=f3.x; kf[13]=f3.y; kf[14]=f3.z; kf[15]=f3.w;
    const float* vp = V + (size_t)(b * SEQ_ + sb + llh * 16) * VSTR_ + kvh * DV_ + lf * 16 + lll;
#pragma unroll
    for (int e = 0; e < 16; ++e) vf[e] = vp[(size_t)e * VSTR_];
  };
  // convert + store the staged tile into LDS buffer `buf` (fragment layout)
  auto store_tile = [&](int buf) {
    unsigned* dk = &kldsu[buf][(lf * 32 + lln) * 8];
    unsigned* dv = &vldsu[buf][(lf * 32 + lln) * 8];
#pragma unroll
    for (int j = 0; j < 8; ++j) dk[j] = pk2(kf[2*j], kf[2*j+1]);
#pragma unroll
    for (int j = 0; j < 8; ++j) dv[j] = pk2(vf[2*j], vf[2*j+1]);
  };

  load_tile(0);
  store_tile(0);
  __syncthreads();

  for (int kt = 0; kt < SEQ_ / 32; ++kt) {
    const int cur = kt & 1;
    const bool more = (kt + 1) < (SEQ_ / 32);
    if (more) load_tile((kt + 1) * 32);   // overlap HBM latency with WMMA below

    // ---- scores: S(16x32) = Q(16x64) . K^T, two 16x16 C tiles ----
    v8f s0 = {}, s1 = {};
#pragma unroll
    for (int c = 0; c < 2; ++c) {
      BFrag kb0, kb1;
      kb0.q[0] = *(const uint4*)&kldsu[cur][((c*2+0)*32 + lane)*8];
      kb0.q[1] = *(const uint4*)&kldsu[cur][((c*2+0)*32 + lane)*8 + 4];
      kb1.q[0] = *(const uint4*)&kldsu[cur][((c*2+1)*32 + lane)*8];
      kb1.q[1] = *(const uint4*)&kldsu[cur][((c*2+1)*32 + lane)*8 + 4];
      s0 = __builtin_amdgcn_wmma_f32_16x16x32_bf16(false, qa[c].v, false, kb0.v, (short)0, s0, false, false);
      s1 = __builtin_amdgcn_wmma_f32_16x16x32_bf16(false, qa[c].v, false, kb1.v, (short)0, s1, false, false);
    }

    // ---- online softmax (C layout: vgpr r -> row r + 8*(lane>=16), col = lane%16) ----
    float p0[8], p1[8], alpha[8];
#pragma unroll
    for (int r = 0; r < 8; ++r) {
      float a  = s0[r] * SCALE_;
      float bb = s1[r] * SCALE_;
      float rx = fmaxf(a, bb);
#pragma unroll
      for (int msk = 8; msk >= 1; msk >>= 1)
        rx = fmaxf(rx, __shfl_xor(rx, msk, 32));    // reduce across 16-col group
      float mn = fmaxf(m[r], rx);
      alpha[r] = __expf(m[r] - mn);
      m[r] = mn;
      float e0 = __expf(a - mn), e1 = __expf(bb - mn);
      p0[r] = e0; p1[r] = e1;
      float rs = e0 + e1;
#pragma unroll
      for (int msk = 8; msk >= 1; msk >>= 1)
        rs += __shfl_xor(rs, msk, 32);
      l[r] = l[r] * alpha[r] + rs;
    }
#pragma unroll
    for (int r = 0; r < 8; ++r) {
      acc0[r] *= alpha[r]; acc1[r] *= alpha[r];
      acc2[r] *= alpha[r]; acc3[r] *= alpha[r];
    }

    // ---- P (C layout) -> A fragment via per-wave LDS relayout ----
#pragma unroll
    for (int r = 0; r < 8; ++r) {
      const int row = r + lh * 8;
      plds[wave][row][ll]      = f2bf(p0[r]);
      plds[wave][row][ll + 16] = f2bf(p1[r]);
    }
    asm volatile("s_wait_dscnt 0" ::: "memory");   // wave-local DS RAW ordering
    BFrag pa;
    {
      const unsigned short* pr = &plds[wave][ll][lh * 16];
      pa.q[0] = *(const uint4*)(pr);        // K {0..7} / {16..23}
      pa.q[1] = *(const uint4*)(pr + 8);    // K {8..15} / {24..31}
    }

    // ---- O(16x64) += P(16x32) . V(32x64): 4 WMMAs ----
    {
      BFrag vb;
      vb.q[0] = *(const uint4*)&vldsu[cur][(0*32 + lane)*8];
      vb.q[1] = *(const uint4*)&vldsu[cur][(0*32 + lane)*8 + 4];
      acc0 = __builtin_amdgcn_wmma_f32_16x16x32_bf16(false, pa.v, false, vb.v, (short)0, acc0, false, false);
      vb.q[0] = *(const uint4*)&vldsu[cur][(1*32 + lane)*8];
      vb.q[1] = *(const uint4*)&vldsu[cur][(1*32 + lane)*8 + 4];
      acc1 = __builtin_amdgcn_wmma_f32_16x16x32_bf16(false, pa.v, false, vb.v, (short)0, acc1, false, false);
      vb.q[0] = *(const uint4*)&vldsu[cur][(2*32 + lane)*8];
      vb.q[1] = *(const uint4*)&vldsu[cur][(2*32 + lane)*8 + 4];
      acc2 = __builtin_amdgcn_wmma_f32_16x16x32_bf16(false, pa.v, false, vb.v, (short)0, acc2, false, false);
      vb.q[0] = *(const uint4*)&vldsu[cur][(3*32 + lane)*8];
      vb.q[1] = *(const uint4*)&vldsu[cur][(3*32 + lane)*8 + 4];
      acc3 = __builtin_amdgcn_wmma_f32_16x16x32_bf16(false, pa.v, false, vb.v, (short)0, acc3, false, false);
    }

    if (more) store_tile(cur ^ 1);   // fill the other buffer for tile kt+1
    __syncthreads();                 // fence + barrier: readers done, buffer published
  }

  // ---- epilogue: normalize and store bf16 O tile ----
  float invl[8];
#pragma unroll
  for (int r = 0; r < 8; ++r) invl[r] = 1.0f / l[r];
#pragma unroll
  for (int r = 0; r < 8; ++r) {
    const int row = q0 + r + lh * 8;
    unsigned short* op = Obf + (size_t)(b * SEQ_ + row) * OSTR_ + h * DV_ + ll;
    op[0]  = f2bf(acc0[r] * invl[r]);
    op[16] = f2bf(acc1[r] * invl[r]);
    op[32] = f2bf(acc2[r] * invl[r]);
    op[48] = f2bf(acc3[r] * invl[r]);
  }
}

// =====================================================================
// Kernel 2: output projection  Out[4096,2048] = Obf . W^T + bias (f32 out)
// grid = (4096/128, 2048/64), block = 256 (8 waves, each 16 rows x 64 cols)
// W k-chunks staged through double-buffered LDS in B-fragment layout,
// shared by all 8 waves; A fragments streamed from global bf16.
// =====================================================================
__global__ __launch_bounds__(256)
void out_proj_kernel(const unsigned short* __restrict__ Obf,
                     const float* __restrict__ W, const float* __restrict__ bias,
                     float* __restrict__ Out)
{
  __shared__ __align__(16) unsigned wldsu[2][4 * 32 * 8];  // 2 x 4 KB

  const int tid  = threadIdx.x;
  const int wave = tid >> 5;
  const int lane = tid & 31;
  const int lh   = lane >> 4, ll = lane & 15;

  const int rb = blockIdx.x * 128 + wave * 16;  // row base in [0, B*S)
  const int cb = blockIdx.y * 64;               // col base in [0, HID)

  v8f acc0 = {}, acc1 = {}, acc2 = {}, acc3 = {};

  // cooperative W loader mapping: 128 fragment-lane slots x 2 halves
  const int slot = tid >> 1;
  const int half = tid & 1;
  const int st = slot >> 5, sln = slot & 31;
  const int sll = sln & 15, slh = sln >> 4;
  const float* wbase = W + (size_t)(cb + st * 16 + sll) * OSTR_ + slh * 8 + half * 16;

  const unsigned short* abase = Obf + (size_t)(rb + ll) * OSTR_ + lh * 8;

  float wf[8];
  auto load_w = [&](int k0) {
    const float* wp = wbase + k0;
    float4 f0 = *(const float4*)(wp + 0);
    float4 f1 = *(const float4*)(wp + 4);
    wf[0]=f0.x; wf[1]=f0.y; wf[2]=f0.z; wf[3]=f0.w;
    wf[4]=f1.x; wf[5]=f1.y; wf[6]=f1.z; wf[7]=f1.w;
  };
  auto store_w = [&](int buf) {
    *(uint4*)&wldsu[buf][slot * 8 + half * 4] =
        make_uint4(pk2(wf[0],wf[1]), pk2(wf[2],wf[3]), pk2(wf[4],wf[5]), pk2(wf[6],wf[7]));
  };

  load_w(0);
  store_w(0);
  __syncthreads();

  for (int k0 = 0; k0 < OSTR_; k0 += 32) {
    const int cur = (k0 >> 5) & 1;
    const bool more = (k0 + 32) < OSTR_;
    if (more) load_w(k0 + 32);   // overlap next W chunk with WMMA below

    BFrag a;
    a.q[0] = *(const uint4*)(abase + k0);
    a.q[1] = *(const uint4*)(abase + k0 + 16);
    __builtin_prefetch(abase + k0 + 32, 0, 1);  // global_prefetch next A chunk

    BFrag wb;
    wb.q[0] = *(const uint4*)&wldsu[cur][(0*32 + lane)*8];
    wb.q[1] = *(const uint4*)&wldsu[cur][(0*32 + lane)*8 + 4];
    acc0 = __builtin_amdgcn_wmma_f32_16x16x32_bf16(false, a.v, false, wb.v, (short)0, acc0, false, false);
    wb.q[0] = *(const uint4*)&wldsu[cur][(1*32 + lane)*8];
    wb.q[1] = *(const uint4*)&wldsu[cur][(1*32 + lane)*8 + 4];
    acc1 = __builtin_amdgcn_wmma_f32_16x16x32_bf16(false, a.v, false, wb.v, (short)0, acc1, false, false);
    wb.q[0] = *(const uint4*)&wldsu[cur][(2*32 + lane)*8];
    wb.q[1] = *(const uint4*)&wldsu[cur][(2*32 + lane)*8 + 4];
    acc2 = __builtin_amdgcn_wmma_f32_16x16x32_bf16(false, a.v, false, wb.v, (short)0, acc2, false, false);
    wb.q[0] = *(const uint4*)&wldsu[cur][(3*32 + lane)*8];
    wb.q[1] = *(const uint4*)&wldsu[cur][(3*32 + lane)*8 + 4];
    acc3 = __builtin_amdgcn_wmma_f32_16x16x32_bf16(false, a.v, false, wb.v, (short)0, acc3, false, false);

    if (more) store_w(cur ^ 1);
    __syncthreads();
  }

  const float b0 = bias[cb + ll];
  const float b1 = bias[cb + ll + 16];
  const float b2 = bias[cb + ll + 32];
  const float b3 = bias[cb + ll + 48];
#pragma unroll
  for (int r = 0; r < 8; ++r) {
    const int row = rb + r + lh * 8;
    float* op = Out + (size_t)row * HID_ + cb + ll;
    op[0]  = acc0[r] + b0;
    op[16] = acc1[r] + b1;
    op[32] = acc2[r] + b2;
    op[48] = acc3[r] + b3;
  }
}

extern "C" void kernel_launch(void* const* d_in, const int* in_sizes, int n_in,
                              void* d_out, int out_size, void* d_ws, size_t ws_size,
                              hipStream_t stream)
{
  const float* Q    = (const float*)d_in[0];
  const float* K    = (const float*)d_in[1];
  const float* V    = (const float*)d_in[2];
  const float* W    = (const float*)d_in[3];
  const float* bias = (const float*)d_in[4];
  float* Out = (float*)d_out;

  // workspace: attention output as bf16 [B*S, HQ*DV] = 2*2048*2048*2B = 16.8 MB
  unsigned short* Obf = (unsigned short*)d_ws;

  dim3 g1(SEQ_ / 64, BATCH_ * HQ_);
  gqa_attn_kernel<<<g1, 128, 0, stream>>>(Q, K, V, Obf);

  dim3 g2((BATCH_ * SEQ_) / 128, HID_ / 64);
  out_proj_kernel<<<g2, 256, 0, stream>>>(Obf, W, bias, Out);
}
